// TransformerEncoder_6047313952823
// MI455X (gfx1250) — compile-verified, matrix operand order checked by reading, not run on hardware
//
#include <hip/hip_runtime.h>

// ---------------------------------------------------------------------------
// Transformer encoder forward, MI455X (gfx1250, wave32, WMMA bf16,
// async global->LDS staging, double-buffered tiles, 64x64 wave tiles).
// ---------------------------------------------------------------------------

typedef unsigned short u16;
typedef __bf16 v16bf __attribute__((ext_vector_type(16)));
typedef float  v8f   __attribute__((ext_vector_type(8)));

#define S_LEN  1024
#define BATCHN 8
#define HDIM   1024
#define FFDIM  4096
#define NLAYER 6
#define MROWS  (S_LEN * BATCHN)   // 8192 token rows

#define BM 128
#define BN 256
#define BK 32
#define LDT 40   // padded LDS row length in bf16 elems (80B, multiple of 16B)

static __device__ __forceinline__ u16 f2bf(float f) {
  union { float f; unsigned u; } c; c.f = f;
  unsigned r = c.u + 0x7FFFu + ((c.u >> 16) & 1u);   // round-to-nearest-even
  return (u16)(r >> 16);
}

// Async copy 16B global -> LDS (CDNA5 GLOBAL_LOAD_ASYNC_TO_LDS_B128).
// Flat pointers to LDS carry the LDS byte offset in their low 32 bits
// (ISA 10.2), which is what VDST wants.
static __device__ __forceinline__ void async_b128(u16* lp, const u16* gp) {
  unsigned off = (unsigned)(unsigned long long)lp;
  asm volatile("global_load_async_to_lds_b128 %0, %1, off"
               :: "v"(off), "v"(gp) : "memory");
}

static __device__ __forceinline__ void wait_async0() {
#if __has_builtin(__builtin_amdgcn_s_wait_asynccnt)
  __builtin_amdgcn_s_wait_asynccnt(0);
#else
  asm volatile("s_wait_asynccnt 0x0" ::: "memory");
#endif
}

// ------------------------------------------- f32 -> bf16 transposed convert
// out[n*rows + k] = bf16(in[k*cols + n]); rows=K, cols=N, both multiples of 32.
__global__ __launch_bounds__(256)
void cvt_transpose_kernel(const float* __restrict__ in, u16* __restrict__ out,
                          int rows, int cols) {
  __shared__ u16 tile[32][33];
  const int k0 = blockIdx.x * 32;
  const int n0 = blockIdx.y * 32;
  const int tx = threadIdx.x & 31;
  const int ty = threadIdx.x >> 5;
  #pragma unroll
  for (int r = ty; r < 32; r += 8)
    tile[r][tx] = f2bf(in[(long)(k0 + r) * cols + (n0 + tx)]);
  __syncthreads();
  #pragma unroll
  for (int r = ty; r < 32; r += 8)
    out[(long)(n0 + r) * rows + (k0 + tx)] = tile[tx][r];
}

// --------------------------------------------- embedding + positional encode
__global__ __launch_bounds__(256)
void embed_pe_kernel(const int* __restrict__ src, const float* __restrict__ emb,
                     float* __restrict__ xf, u16* __restrict__ xb) {
  const int row = blockIdx.x;          // s*B + b
  const int s = row / BATCHN;
  const int tok = src[row];
  const float* er = emb + (long)tok * HDIM;
  for (int h = threadIdx.x; h < HDIM; h += 256) {
    int j2 = h & ~1;
    float div = __expf((float)j2 * (-9.210340371976184f / (float)HDIM));
    float ang = (float)s * div;
    float pe = (h & 1) ? __cosf(ang) : __sinf(ang);
    float v = er[h] * 32.0f + pe;      // scale = sqrt(H) = 32
    xf[(long)row * HDIM + h] = v;
    xb[(long)row * HDIM + h] = f2bf(v);
  }
}

// --------------------------------------------------------- WMMA bf16 GEMM
// Block tile 128x256, 8 waves, each wave 64x64 (4x4 WMMA tiles / K-step).
// TRANSB=1: B element (k,n) = Bb[n*ldb + k]  (pure async B128 staging)
// TRANSB=0: B row-major [K][N]               (ds-store transpose staging)
template <int TRANSB>
__global__ __launch_bounds__(256)
void gemm_wmma_kernel(const u16* __restrict__ A, long lda, long aBatch,
                      const u16* __restrict__ Bg, long ldb, long bBatch,
                      const float* __restrict__ bias,
                      float* __restrict__ Cf, u16* __restrict__ Cb,
                      long ldc, long cBatch, int K, float alpha, int relu) {
  __shared__ u16 As[2][BM * LDT];
  __shared__ u16 Bs[2][BN * LDT];

  const int tid = threadIdx.x;
  const long m0 = (long)blockIdx.x * BM;
  const long n0 = (long)blockIdx.y * BN;
  const int bz = blockIdx.z;
  const u16* Ab = A + (long)bz * aBatch;
  const u16* Bb = Bg + (long)bz * bBatch;

  const int lane = tid & 31;
  const int wave = tid >> 5;           // 8 waves: 2 x 4 layout
  const int wm = (wave >> 2) * 64;     // 0 or 64
  const int wn = (wave & 3) * 64;      // 0,64,128,192
  const int lh = lane & 15;
  const int g  = lane >> 4;            // half-wave group (ISA fragment layout)

  const int sm  = tid >> 1;            // A staging: 0..127 row
  const int skh = (tid & 1) << 4;      // 0 or 16
  const int bk  = tid >> 3;            // non-trans B: 0..31 k row
  const int bnb = (tid & 7) << 5;      // non-trans B: 0..224 n base

  v8f zero;
  #pragma unroll
  for (int e = 0; e < 8; ++e) zero[e] = 0.f;
  v8f acc[4][4];
  #pragma unroll
  for (int i = 0; i < 4; ++i)
    #pragma unroll
    for (int j = 0; j < 4; ++j) acc[i][j] = zero;

  auto stage = [&](int buf, int k0) {
    { // A tile: async, two contiguous 16B chunks per thread (128 rows x 32 k)
      u16* l = &As[buf][sm * LDT + skh];
      const u16* gp = Ab + (m0 + sm) * lda + (k0 + skh);
      async_b128(l, gp);
      async_b128(l + 16, gp + 16);
    }
    if (TRANSB) { // B tile: one row per thread (256 rows x 32 k), async
      u16* l = &Bs[buf][tid * LDT];
      const u16* gp = Bb + (n0 + tid) * ldb + k0;
      async_b128(l, gp);
      async_b128(l + 16, gp + 16);
    } else {      // transpose B[K][N] into Bs[n][k]: 32 contiguous n per thread
      const u16* gp = Bb + (long)(k0 + bk) * ldb + (n0 + bnb);
      uint4 d[4];
      #pragma unroll
      for (int c = 0; c < 4; ++c) d[c] = *(const uint4*)(gp + c * 8);
      const u16* e = (const u16*)d;
      #pragma unroll
      for (int j = 0; j < 32; ++j)
        Bs[buf][(bnb + j) * LDT + bk] = e[j];
    }
  };

  stage(0, 0);                         // prologue
  int buf = 0;
  for (int k0 = 0; k0 < K; k0 += BK) {
    wait_async0();                     // this wave's async tile landed
    __syncthreads();                   // all waves' tiles landed (incl. dscnt)
    if (k0 + BK < K) stage(buf ^ 1, k0 + BK);
    if (k0 + 2 * BK < K) {             // L2 warm for tile after next
      __builtin_prefetch(Ab + (m0 + sm) * lda + (k0 + 2 * BK + skh), 0, 2);
      if (TRANSB)
        __builtin_prefetch(Bb + (n0 + tid) * ldb + (k0 + 2 * BK), 0, 2);
      else
        __builtin_prefetch(Bb + (long)(k0 + 2 * BK + bk) * ldb + (n0 + bnb), 0, 2);
    }

    // Fragment loads per CDNA5 16-bit A/B layout: lane (g,lh) reads two
    // contiguous 16B chunks at k = g*8 and k = 16 + g*8.
    union FragU { uint4 u[2]; v16bf v; };
    v16bf afr[4], bfr[4];
    #pragma unroll
    for (int tm = 0; tm < 4; ++tm) {
      const u16* p = &As[buf][(wm + tm * 16 + lh) * LDT + g * 8];
      FragU f; f.u[0] = *(const uint4*)p; f.u[1] = *(const uint4*)(p + 16);
      afr[tm] = f.v;
    }
    #pragma unroll
    for (int tn = 0; tn < 4; ++tn) {
      const u16* p = &Bs[buf][(wn + tn * 16 + lh) * LDT + g * 8];
      FragU f; f.u[0] = *(const uint4*)p; f.u[1] = *(const uint4*)(p + 16);
      bfr[tn] = f.v;
    }
    #pragma unroll
    for (int tm = 0; tm < 4; ++tm)
      #pragma unroll
      for (int tn = 0; tn < 4; ++tn)
        acc[tm][tn] = __builtin_amdgcn_wmma_f32_16x16x32_bf16(
            false, afr[tm], false, bfr[tn], (short)0, acc[tm][tn], false, false);
    buf ^= 1;
  }

  // Epilogue: C/D layout — lane lh = column, VGPR i -> row g*8 + i.
  #pragma unroll
  for (int tn = 0; tn < 4; ++tn) {
    const long col = n0 + wn + tn * 16 + lh;
    const float bv = bias ? bias[col] : 0.f;
    #pragma unroll
    for (int tm = 0; tm < 4; ++tm) {
      #pragma unroll
      for (int i = 0; i < 8; ++i) {
        const long rowm = m0 + wm + tm * 16 + g * 8 + i;
        float v = (acc[tm][tn][i] + bv) * alpha;
        if (relu) v = fmaxf(v, 0.f);
        const long idx = (long)bz * cBatch + rowm * ldc + col;
        if (Cf) Cf[idx] = v;
        if (Cb) Cb[idx] = f2bf(v);
      }
    }
  }
}

// ------------------------------------------------------- masked row softmax
__global__ __launch_bounds__(256)
void softmax_mask_kernel(const float* __restrict__ scores,
                         const unsigned char* __restrict__ mask,
                         u16* __restrict__ P) {
  const int b = blockIdx.x >> 10;
  const int q = blockIdx.x & (S_LEN - 1);
  const float* row = scores + ((long)b * S_LEN + q) * S_LEN;
  u16* prow = P + ((long)b * S_LEN + q) * S_LEN;
  __shared__ float red[8];
  const int t = threadIdx.x, lane = t & 31, wave = t >> 5;

  float vals[4];
  float mx = -3.0e38f;
  #pragma unroll
  for (int j = 0; j < 4; ++j) {
    int k = t + j * 256;
    float v = row[k];
    if (!mask[(long)k * BATCHN + b]) v = -1e9f;
    vals[j] = v;
    mx = fmaxf(mx, v);
  }
  #pragma unroll
  for (int off = 16; off; off >>= 1) mx = fmaxf(mx, __shfl_xor(mx, off, 32));
  if (lane == 0) red[wave] = mx;
  __syncthreads();
  mx = red[0];
  #pragma unroll
  for (int j = 1; j < 8; ++j) mx = fmaxf(mx, red[j]);
  __syncthreads();

  float sum = 0.f;
  #pragma unroll
  for (int j = 0; j < 4; ++j) { vals[j] = __expf(vals[j] - mx); sum += vals[j]; }
  #pragma unroll
  for (int off = 16; off; off >>= 1) sum += __shfl_xor(sum, off, 32);
  if (lane == 0) red[wave] = sum;
  __syncthreads();
  sum = 0.f;
  #pragma unroll
  for (int j = 0; j < 8; ++j) sum += red[j];
  const float inv = 1.0f / sum;
  #pragma unroll
  for (int j = 0; j < 4; ++j) prow[t + j * 256] = f2bf(vals[j] * inv);
}

// -------------------------------------------------- (residual +) layer norm
__global__ __launch_bounds__(256)
void add_ln_kernel(const float* __restrict__ xin, const float* __restrict__ resid,
                   const float* __restrict__ gamma, const float* __restrict__ beta,
                   float* __restrict__ xout, u16* __restrict__ bout) {
  const long row = blockIdx.x;
  const float* xr = xin + row * HDIM;
  const float* rr = resid ? resid + row * HDIM : nullptr;
  __shared__ float red[8];
  const int t = threadIdx.x, lane = t & 31, wave = t >> 5;

  float vals[4];
  float s = 0.f;
  #pragma unroll
  for (int j = 0; j < 4; ++j) {
    int h = t + j * 256;
    float v = xr[h] + (rr ? rr[h] : 0.f);
    vals[j] = v;
    s += v;
  }
  #pragma unroll
  for (int off = 16; off; off >>= 1) s += __shfl_xor(s, off, 32);
  if (lane == 0) red[wave] = s;
  __syncthreads();
  s = 0.f;
  #pragma unroll
  for (int j = 0; j < 8; ++j) s += red[j];
  __syncthreads();
  const float mean = s * (1.0f / (float)HDIM);

  float vs = 0.f;
  #pragma unroll
  for (int j = 0; j < 4; ++j) { float d = vals[j] - mean; vs += d * d; }
  #pragma unroll
  for (int off = 16; off; off >>= 1) vs += __shfl_xor(vs, off, 32);
  if (lane == 0) red[wave] = vs;
  __syncthreads();
  vs = 0.f;
  #pragma unroll
  for (int j = 0; j < 8; ++j) vs += red[j];
  const float inv = rsqrtf(vs * (1.0f / (float)HDIM) + 1e-5f);

  #pragma unroll
  for (int j = 0; j < 4; ++j) {
    int h = t + j * 256;
    float y = (vals[j] - mean) * inv * gamma[h] + beta[h];
    xout[row * HDIM + h] = y;
    if (bout) bout[row * HDIM + h] = f2bf(y);
  }
}

// ---------------------------------------------------------------------------
static inline void cvtT(const float* src, u16* dst, int rows, int cols,
                        hipStream_t s) {
  dim3 g(rows / 32, cols / 32, 1);
  cvt_transpose_kernel<<<g, 256, 0, s>>>(src, dst, rows, cols);
}

extern "C" void kernel_launch(void* const* d_in, const int* in_sizes, int n_in,
                              void* d_out, int out_size, void* d_ws, size_t ws_size,
                              hipStream_t stream) {
  (void)in_sizes; (void)n_in; (void)out_size; (void)ws_size;

  const int*           src  = (const int*)d_in[0];
  const unsigned char* mask = (const unsigned char*)d_in[1];
  const float* emb = (const float*)d_in[2];
  const float* Wq  = (const float*)d_in[3];
  const float* bq  = (const float*)d_in[4];
  const float* Wk  = (const float*)d_in[5];
  const float* bk  = (const float*)d_in[6];
  const float* Wv  = (const float*)d_in[7];
  const float* bv  = (const float*)d_in[8];
  const float* Wo  = (const float*)d_in[9];
  const float* bo  = (const float*)d_in[10];
  const float* W1  = (const float*)d_in[11];
  const float* b1  = (const float*)d_in[12];
  const float* W2  = (const float*)d_in[13];
  const float* b2  = (const float*)d_in[14];
  const float* g1  = (const float*)d_in[15];
  const float* be1 = (const float*)d_in[16];
  const float* g2  = (const float*)d_in[17];
  const float* be2 = (const float*)d_in[18];
  const float* gf  = (const float*)d_in[19];
  const float* bf  = (const float*)d_in[20];

  constexpr size_t MB = 1u << 20;
  char* ws = (char*)d_ws;
  float* xf    = (float*)(ws + 0);           // 32 MB  x (f32)
  u16*   xb    = (u16*)  (ws + 32  * MB);    // 16 MB  x (bf16)
  u16*   qb    = (u16*)  (ws + 48  * MB);    // 16 MB
  u16*   kb    = (u16*)  (ws + 64  * MB);    // 16 MB
  u16*   vb    = (u16*)  (ws + 80  * MB);    // 16 MB
  u16*   ctxb  = (u16*)  (ws + 96  * MB);    // 16 MB
  u16*   pb    = (u16*)  (ws + 112 * MB);    // 16 MB  softmax probs (bf16)
  float* scf   = (float*)(ws + 128 * MB);    // 32 MB  attention scores
  float* tmpf  = (float*)(ws + 160 * MB);    // 32 MB  residual branch
  u16*   ff1b  = (u16*)  (ws + 192 * MB);    // 64 MB  ReLU(ff1) (bf16)
  u16*   wb    = (u16*)  (ws + 256 * MB);    // 24 MB  per-layer bf16 W^T
  u16* wqb = wb;                             // all stored transposed [N][K]
  u16* wkb = wb + (1u << 20);
  u16* wvb = wb + (2u << 20);
  u16* wob = wb + (3u << 20);
  u16* w1b = wb + (4u << 20);                // W1^T [FF][H]
  u16* w2b = wb + (8u << 20);                // W2^T [H][FF]

  const long HH = (long)HDIM * HDIM;
  const long HF = (long)HDIM * FFDIM;
  const long SS = (long)S_LEN * S_LEN;

  embed_pe_kernel<<<MROWS, 256, 0, stream>>>(src, emb, xf, xb);

  const dim3 gProj(MROWS / BM, HDIM / BN, 1);        // 64 x 4
  const dim3 gAttn(S_LEN / BM, S_LEN / BN, BATCHN);  //  8 x 4 x 8
  const dim3 gCtx (S_LEN / BM, HDIM / BN, BATCHN);   //  8 x 4 x 8
  const dim3 gFF1 (MROWS / BM, FFDIM / BN, 1);       // 64 x 16

  for (int l = 0; l < NLAYER; ++l) {
    cvtT(Wq + l * HH, wqb, HDIM, HDIM, stream);
    cvtT(Wk + l * HH, wkb, HDIM, HDIM, stream);
    cvtT(Wv + l * HH, wvb, HDIM, HDIM, stream);
    cvtT(Wo + l * HH, wob, HDIM, HDIM, stream);
    cvtT(W1 + l * HF, w1b, HDIM, FFDIM, stream);   // -> [FF][H]
    cvtT(W2 + l * HF, w2b, FFDIM, HDIM, stream);   // -> [H][FF]

    // q = (x Wq + bq)/32 ; k = x Wk + bk ; v = x Wv + bv   (W^T, async path)
    gemm_wmma_kernel<1><<<gProj, 256, 0, stream>>>(xb, HDIM, 0, wqb, HDIM, 0,
        bq + l * HDIM, nullptr, qb, HDIM, 0, HDIM, 1.0f / 32.0f, 0);
    gemm_wmma_kernel<1><<<gProj, 256, 0, stream>>>(xb, HDIM, 0, wkb, HDIM, 0,
        bk + l * HDIM, nullptr, kb, HDIM, 0, HDIM, 1.0f, 0);
    gemm_wmma_kernel<1><<<gProj, 256, 0, stream>>>(xb, HDIM, 0, wvb, HDIM, 0,
        bv + l * HDIM, nullptr, vb, HDIM, 0, HDIM, 1.0f, 0);

    // scores[b] = Q_b K_b^T  (rows strided by B*H, batch offset b*H)
    gemm_wmma_kernel<1><<<gAttn, 256, 0, stream>>>(qb, MROWS, HDIM,
        kb, MROWS, HDIM, nullptr, scf, nullptr, S_LEN, SS, HDIM, 1.0f, 0);

    softmax_mask_kernel<<<BATCHN * S_LEN, 256, 0, stream>>>(scf, mask, pb);

    // ctx[b] = P_b V_b ; scatter rows back to (s*B+b)*H layout
    gemm_wmma_kernel<0><<<gCtx, 256, 0, stream>>>(pb, S_LEN, SS,
        vb, MROWS, HDIM, nullptr, nullptr, ctxb, MROWS, HDIM, S_LEN, 1.0f, 0);

    // src2 = ctx Wo + bo ; x = LN(x + src2)
    gemm_wmma_kernel<1><<<gProj, 256, 0, stream>>>(ctxb, HDIM, 0, wob, HDIM, 0,
        bo + l * HDIM, tmpf, nullptr, HDIM, 0, HDIM, 1.0f, 0);
    add_ln_kernel<<<MROWS, 256, 0, stream>>>(xf, tmpf,
        g1 + l * HDIM, be1 + l * HDIM, xf, xb);

    // ff = relu(x W1 + b1) ; ff2 = ff W2 + b2 ; x = LN(x + ff2)
    gemm_wmma_kernel<1><<<gFF1, 256, 0, stream>>>(xb, HDIM, 0, w1b, HDIM, 0,
        b1 + l * FFDIM, nullptr, ff1b, FFDIM, 0, HDIM, 1.0f, 1);
    gemm_wmma_kernel<1><<<gProj, 256, 0, stream>>>(ff1b, FFDIM, 0, w2b, FFDIM, 0,
        b2 + l * HDIM, tmpf, nullptr, HDIM, 0, FFDIM, 1.0f, 0);
    add_ln_kernel<<<MROWS, 256, 0, stream>>>(xf, tmpf,
        g2 + l * HDIM, be2 + l * HDIM, xf, xb);
  }

  add_ln_kernel<<<MROWS, 256, 0, stream>>>(xf, nullptr, gf, bf,
                                           (float*)d_out, nullptr);
}
